// GraphFormerEncoder_23587960389879
// MI455X (gfx1250) — compile-verified, matrix operand order checked by reading, not run on hardware
//
#include <hip/hip_runtime.h>
#include <hip/hip_bf16.h>
#include <math.h>

// ---------------------------------------------------------------------------
// Problem constants (match reference)
// ---------------------------------------------------------------------------
#define NN   2048
#define TT   8
#define BB   16
#define NPG  128
#define HH   256
#define NHD  2
#define DHD  128
#define LL   3

typedef _Float16 half16 __attribute__((ext_vector_type(16)));
typedef _Float16 half8  __attribute__((ext_vector_type(8)));
typedef float    float8 __attribute__((ext_vector_type(8)));

// ---------------------------------------------------------------------------
// WMMA helpers (CDNA5: V_WMMA_F32_16X16X32_F16, wave32)
// ---------------------------------------------------------------------------
static __device__ __forceinline__ float8 wmma_fp16(half16 a, half16 b, float8 c) {
    return __builtin_amdgcn_wmma_f32_16x16x32_f16(false, a, false, b, (short)0, c, false, false);
}

// Fragment-native ("swizzled") LDS layout for one 16x32 f16 chunk (512 elems).
// Element (rc, k), rc = row (A) or col (B), k = 0..31, lives at:
//   lane = (rc & 15) + 16*((k >> 3) & 1)
//   idx  = (k & 7)   +  8*((k >> 4) & 1)
//   chunk[lane*16 + idx]
// => a lane's whole fragment is 32 contiguous bytes at chunk + lane*16,
//    loadable as 2x ds_load_b128. For an 8-aligned k-octet the destination
//    indices are contiguous -> staging uses ds_store_b128.
static __device__ __forceinline__ int swz_idx(int rc, int k) {
    int lane = (rc & 15) + (((k >> 3) & 1) << 4);
    int idx  = (k & 7) + (((k >> 4) & 1) << 3);
    return lane * 16 + idx;
}

static __device__ __forceinline__ half16 load_frag(const _Float16* chunk, int lane) {
    return *(const half16*)(chunk + lane * 16);
}

// convert 8 consecutive f32 -> half8
static __device__ __forceinline__ half8 cvt8(const float* __restrict__ src) {
    float4 a = *(const float4*)(src);
    float4 b = *(const float4*)(src + 4);
    half8 r;
    r[0] = (_Float16)a.x; r[1] = (_Float16)a.y; r[2] = (_Float16)a.z; r[3] = (_Float16)a.w;
    r[4] = (_Float16)b.x; r[5] = (_Float16)b.y; r[6] = (_Float16)b.z; r[7] = (_Float16)b.w;
    return r;
}

// convert 8 strided f32 (stride in floats) -> half8
static __device__ __forceinline__ half8 cvt8s(const float* __restrict__ src, int stride) {
    half8 r;
#pragma unroll
    for (int i = 0; i < 8; ++i) r[i] = (_Float16)src[i * stride];
    return r;
}

static __device__ __forceinline__ float gelu_f(float x) {
    return 0.5f * x * (1.0f + erff(x * 0.70710678118654752f));
}

// Block-wide LayerNorm over H=256 (one row per block, 256 threads)
static __device__ __forceinline__ float block_ln(float v, float* red, const float* g,
                                                 const float* b, int h) {
    red[h] = v;
    __syncthreads();
    for (int s = 128; s > 0; s >>= 1) { if (h < s) red[h] += red[h + s]; __syncthreads(); }
    float mean = red[0] * (1.0f / 256.0f);
    __syncthreads();
    float d = v - mean;
    red[h] = d * d;
    __syncthreads();
    for (int s = 128; s > 0; s >>= 1) { if (h < s) red[h] += red[h + s]; __syncthreads(); }
    float var = red[0] * (1.0f / 256.0f);
    __syncthreads();
    return d * rsqrtf(var + 1e-5f) * g[h] + b[h];
}

// ---------------------------------------------------------------------------
// Kernel 1: broad_features [B,T,5] @ Wp [5,H] + bp  -> broad [B*T, H]
// ---------------------------------------------------------------------------
__global__ __launch_bounds__(256) void broad_proj_kernel(const float* __restrict__ bf,
                                                         const float* __restrict__ Wp,
                                                         const float* __restrict__ bp,
                                                         float* __restrict__ outp) {
    int row = blockIdx.x;
    int h = threadIdx.x;
    float v = bp[h];
#pragma unroll
    for (int k = 0; k < 5; ++k) v += bf[row * 5 + k] * Wp[k * HH + h];
    outp[(size_t)row * HH + h] = v;
}

// ---------------------------------------------------------------------------
// Kernel 2: node embed + in_projector stage 1 (LN -> GELU), output [T,N,H]
// ---------------------------------------------------------------------------
__global__ __launch_bounds__(256) void embed_ln_gelu_kernel(const float* __restrict__ nodes,
                                                            const float* __restrict__ Wn,
                                                            const float* __restrict__ bn,
                                                            const float* __restrict__ broad,
                                                            const float* __restrict__ types_tab,
                                                            const int* __restrict__ node_types,
                                                            const int* __restrict__ graph_ids,
                                                            const float* __restrict__ g1,
                                                            const float* __restrict__ b1,
                                                            float* __restrict__ X) {
    __shared__ float red[256];
    int bid = blockIdx.x;                 // n*T + t
    int n = bid >> 3, t = bid & 7;
    int h = threadIdx.x;
    float v = bn[h];
#pragma unroll
    for (int k = 0; k < 10; ++k) v += nodes[(size_t)bid * 10 + k] * Wn[k * HH + h];
    v += broad[(size_t)(graph_ids[n] * TT + t) * HH + h];
    v += types_tab[node_types[n] * HH + h];
    float y = gelu_f(block_ln(v, red, g1, b1, h));
    X[((size_t)t * NN + n) * HH + h] = y;
}

// ---------------------------------------------------------------------------
// Kernel 3: WMMA GEMM  C[M,256] = A[M,256] @ W[256,256] + bias (opt GELU)
//  block = 256 thr (8 waves), tile 128x64, swizzled f16 LDS staging.
//  sA: 8 strips (16 rows) x 512;  sB: 4 col-tiles (16 cols) x 512
// ---------------------------------------------------------------------------
__global__ __launch_bounds__(256) void gemm_wmma_kernel(const float* __restrict__ A,
                                                        const float* __restrict__ W,
                                                        const float* __restrict__ bias,
                                                        float* __restrict__ C,
                                                        int M, int act) {
    __shared__ __align__(32) _Float16 sA[8 * 512];
    __shared__ __align__(32) _Float16 sB[4 * 512];
    const int tid = threadIdx.x, wave = tid >> 5, lane = tid & 31;
    const int m0 = blockIdx.x * 128, n0 = blockIdx.y * 64;

    float8 acc[4] = {};
    for (int kt = 0; kt < 8; ++kt) {
        __syncthreads();
        // stage A: 512 octet-tasks (row, k-octet); 2 per thread; wide ld/st
#pragma unroll
        for (int q = 0; q < 2; ++q) {
            int j = tid + q * 256;
            int row = j >> 2, k0 = (j & 3) * 8;
            half8 v = cvt8(&A[(size_t)(m0 + row) * HH + kt * 32 + k0]);
            *(half8*)(sA + (row >> 4) * 512 + swz_idx(row & 15, k0)) = v;
        }
        // stage B: 256 octet-tasks (k-octet, col); 1 per thread (strided reads)
        {
            int oct = tid >> 6, col = tid & 63;
            int k0 = oct * 8;
            half8 v = cvt8s(&W[(size_t)(kt * 32 + k0) * HH + n0 + col], HH);
            *(half8*)(sB + (col >> 4) * 512 + swz_idx(col & 15, k0)) = v;
        }
        __syncthreads();
        half16 a = load_frag(sA + wave * 512, lane);
#pragma unroll
        for (int j = 0; j < 4; ++j)
            acc[j] = wmma_fp16(a, load_frag(sB + j * 512, lane), acc[j]);
    }
#pragma unroll
    for (int j = 0; j < 4; ++j) {
        int col = n0 + j * 16 + (lane & 15);
        float bv = bias[col];
#pragma unroll
        for (int r = 0; r < 8; ++r) {
            int row = m0 + wave * 16 + r + ((lane >> 4) << 3);
            float v = acc[j][r] + bv;
            if (act) v = gelu_f(v);
            C[(size_t)row * HH + col] = v;
        }
    }
}

// ---------------------------------------------------------------------------
// Kernel 4: row LayerNorm (opt residual add, opt GELU)
// ---------------------------------------------------------------------------
__global__ __launch_bounds__(256) void ln_kernel(const float* __restrict__ x,
                                                 const float* __restrict__ res,
                                                 const float* __restrict__ g,
                                                 const float* __restrict__ b,
                                                 float* __restrict__ out, int gelu) {
    __shared__ float red[256];
    int row = blockIdx.x, h = threadIdx.x;
    float v = x[(size_t)row * HH + h];
    if (res) v += res[(size_t)row * HH + h];
    float y = block_ln(v, red, g, b, h);
    if (gelu) y = gelu_f(y);
    out[(size_t)row * HH + h] = y;
}

// ---------------------------------------------------------------------------
// Kernel 5: fused flash attention with spatial bias.
//  grid = (N/64, NH, T); block = 128 thr (4 waves); each wave: 16 query rows.
//  All LDS tiles stored fragment-native (swizzled):
//   sK: [4 d-chunks][2 key-subtiles][512]   (K^T B-fragments for S)
//   sV: [8 d-col-tiles][512]                (V  B-fragments for P@V)
//   sQ: [wave][4 d-chunks][512]             (Q  A-fragments)
//   sP: [wave][512]                         (P  A-fragment, C->A via LDS)
// ---------------------------------------------------------------------------
__global__ __launch_bounds__(128) void flash_attn_kernel(const float* __restrict__ Q,
                                                         const float* __restrict__ K,
                                                         const float* __restrict__ V,
                                                         const int* __restrict__ dist,
                                                         const float* __restrict__ spat,
                                                         float* __restrict__ O) {
    const int t = blockIdx.z, h = blockIdx.y;
    const int tid = threadIdx.x, wave = tid >> 5, lane = tid & 31;
    const int q0 = (blockIdx.x * 4 + wave) * 16;

    __shared__ __align__(32) _Float16 sK[4 * 2 * 512];
    __shared__ __align__(32) _Float16 sV[8 * 512];
    __shared__ __align__(32) _Float16 sQ[4 * 4 * 512];
    __shared__ __align__(32) _Float16 sP[4 * 512];
    __shared__ float sSpat[16];

    if (tid < 14) sSpat[tid] = spat[tid];

    // stage this wave's Q tile into swizzled A-fragment form (8 octets/lane)
#pragma unroll
    for (int q = 0; q < 8; ++q) {
        int j = lane + q * 32;               // 256 tasks: (row, d-octet)
        int row = j >> 4, d0 = (j & 15) * 8;
        half8 v = cvt8(&Q[((size_t)t * NN + q0 + row) * HH + h * DHD + d0]);
        int c = d0 >> 5, kd0 = d0 & 31;
        *(half8*)(sQ + wave * 2048 + c * 512 + swz_idx(row, kd0)) = v;
    }
    __syncthreads();

    half16 qf[4];
#pragma unroll
    for (int c = 0; c < 4; ++c) qf[c] = load_frag(sQ + wave * 2048 + c * 512, lane);

    float8 acc[8] = {};
    float m[8], ssum[8];
#pragma unroll
    for (int r = 0; r < 8; ++r) { m[r] = -INFINITY; ssum[r] = 0.0f; }

    const float scale = 0.08838834764831845f;  // 1/sqrt(128)
    const int colk = lane & 15;
    const int rowoff = (lane >> 4) << 3;

    for (int kt = 0; kt < NN / 32; ++kt) {
        __syncthreads();
        // stage K tile: 512 tasks (key, d-octet); 4 per thread; wide ld/st
#pragma unroll
        for (int q = 0; q < 4; ++q) {
            int j = tid + q * 128;
            int key = j >> 4, d0 = (j & 15) * 8;
            half8 v = cvt8(&K[((size_t)t * NN + kt * 32 + key) * HH + h * DHD + d0]);
            int c = d0 >> 5, kd0 = d0 & 31;
            *(half8*)(sK + c * 1024 + (key >> 4) * 512 + swz_idx(key & 15, kd0)) = v;
        }
        // stage V tile: 512 tasks (kk-octet, d-col); 4 per thread (strided reads)
#pragma unroll
        for (int q = 0; q < 4; ++q) {
            int j = tid + q * 128;
            int oct = j >> 7, col = j & 127;
            int kk0 = oct * 8;
            half8 v = cvt8s(&V[((size_t)t * NN + kt * 32 + kk0) * HH + h * DHD + col], HH);
            *(half8*)(sV + (col >> 4) * 512 + swz_idx(col & 15, kk0)) = v;
        }
        __syncthreads();

        // S = Q @ K^T  (16x32 = two 16x16 column tiles, d = 4 chunks of 32)
        float8 s0 = {}, s1 = {};
#pragma unroll
        for (int c = 0; c < 4; ++c) {
            s0 = wmma_fp16(qf[c], load_frag(sK + c * 1024, lane), s0);
            s1 = wmma_fp16(qf[c], load_frag(sK + c * 1024 + 512, lane), s1);
        }

        // spatial bias + online softmax (per C-layout row r)
#pragma unroll
        for (int r = 0; r < 8; ++r) {
            int qg = q0 + r + rowoff;
            int k0g = kt * 32 + colk;
            int d0 = dist[(size_t)qg * NN + k0g];
            int d1 = dist[(size_t)qg * NN + k0g + 16];
            int i0 = (d0 < 0) ? 0 : ((d0 >= 5) ? 6 : d0 + 1);
            int i1 = (d1 < 0) ? 0 : ((d1 >= 5) ? 6 : d1 + 1);
            float v0 = s0[r] * scale + (i0 ? sSpat[i0 * 2 + h] : 0.0f);
            float v1 = s1[r] * scale + (i1 ? sSpat[i1 * 2 + h] : 0.0f);

            float mloc = fmaxf(v0, v1);
#pragma unroll
            for (int off = 1; off < 16; off <<= 1)
                mloc = fmaxf(mloc, __shfl_xor(mloc, off, 16));
            float mn = fmaxf(m[r], mloc);
            float al = expf(m[r] - mn);
            float p0 = expf(v0 - mn), p1 = expf(v1 - mn);
            float ps = p0 + p1;
#pragma unroll
            for (int off = 1; off < 16; off <<= 1) ps += __shfl_xor(ps, off, 16);
            ssum[r] = ssum[r] * al + ps;
            m[r] = mn;
#pragma unroll
            for (int j = 0; j < 8; ++j) acc[j][r] *= al;
            int qloc = r + rowoff;
            sP[wave * 512 + swz_idx(qloc, colk)]      = (_Float16)p0;
            sP[wave * 512 + swz_idx(qloc, colk + 16)] = (_Float16)p1;
        }

        // wave-private C->A layout round-trip through LDS: wait for ds stores
        asm volatile("s_wait_dscnt 0" ::: "memory");

        half16 pf = load_frag(sP + wave * 512, lane);
#pragma unroll
        for (int j = 0; j < 8; ++j)
            acc[j] = wmma_fp16(pf, load_frag(sV + j * 512, lane), acc[j]);
    }

    // normalize and write O (layout [T,N,H])
#pragma unroll
    for (int r = 0; r < 8; ++r) {
        float inv = 1.0f / ssum[r];
        int qg = q0 + r + rowoff;
#pragma unroll
        for (int j = 0; j < 8; ++j)
            O[((size_t)t * NN + qg) * HH + h * DHD + j * 16 + colk] = acc[j][r] * inv;
    }
}

// ---------------------------------------------------------------------------
// Kernel 6: segment-mean pooling  pooled[b*T+t, h] = mean_n X[t, b*128+n, h]
// ---------------------------------------------------------------------------
__global__ __launch_bounds__(256) void pool_kernel(const float* __restrict__ X,
                                                   float* __restrict__ pooled) {
    int gid = blockIdx.x * 256 + threadIdx.x;
    if (gid >= BB * TT * HH) return;
    int h = gid & (HH - 1);
    int bt = gid / HH;
    int b = bt / TT, t = bt % TT;
    float s = 0.0f;
    for (int n = 0; n < NPG; ++n)
        s += X[((size_t)t * NN + b * NPG + n) * HH + h];
    pooled[gid] = s * (1.0f / (float)NPG);
}

// ---------------------------------------------------------------------------
// Host-side launch sequence
// ---------------------------------------------------------------------------
extern "C" void kernel_launch(void* const* d_in, const int* in_sizes, int n_in,
                              void* d_out, int out_size, void* d_ws, size_t ws_size,
                              hipStream_t stream) {
    const float* nodes     = (const float*)d_in[0];
    const float* broadf    = (const float*)d_in[1];
    const int*   ntypes    = (const int*)d_in[2];
    const int*   gids      = (const int*)d_in[3];
    const int*   dist      = (const int*)d_in[4];
    const float* Wn        = (const float*)d_in[5];
    const float* bn        = (const float*)d_in[6];
    const float* Wp        = (const float*)d_in[7];
    const float* bp        = (const float*)d_in[8];
    const float* types_tab = (const float*)d_in[9];
    const float* g_ip1     = (const float*)d_in[10];
    const float* b_ip1     = (const float*)d_in[11];
    const float* W_ip      = (const float*)d_in[12];
    const float* b_ip      = (const float*)d_in[13];
    const float* g_ip2     = (const float*)d_in[14];
    const float* b_ip2     = (const float*)d_in[15];
    const float* spat      = (const float*)d_in[16];
    const float* Wq        = (const float*)d_in[17];
    const float* bq        = (const float*)d_in[18];
    const float* Wk        = (const float*)d_in[19];
    const float* bk        = (const float*)d_in[20];
    const float* Wv        = (const float*)d_in[21];
    const float* bv        = (const float*)d_in[22];
    const float* Wo        = (const float*)d_in[23];
    const float* bo        = (const float*)d_in[24];
    const float* ln1g      = (const float*)d_in[25];
    const float* ln1b      = (const float*)d_in[26];
    const float* W1f       = (const float*)d_in[27];
    const float* b1f       = (const float*)d_in[28];
    const float* W2f       = (const float*)d_in[29];
    const float* b2f       = (const float*)d_in[30];
    const float* ln2g      = (const float*)d_in[31];
    const float* ln2b      = (const float*)d_in[32];
    const float* Wout      = (const float*)d_in[33];
    const float* bout      = (const float*)d_in[34];
    float* out = (float*)d_out;

    const size_t S = (size_t)TT * NN * HH;      // 4,194,304 floats per activation buf
    float* X      = (float*)d_ws;
    float* buf0   = X + S;
    float* buf1   = X + 2 * S;
    float* buf2   = X + 3 * S;
    float* buf3   = X + 4 * S;
    float* broadb = X + 5 * S;                  // [B*T, H]
    float* pooled = broadb + (size_t)BB * TT * HH;

    const int MROWS = TT * NN;                  // 16384
    dim3 gemmGrid(MROWS / 128, HH / 64);
    dim3 gemmBlk(256);
    dim3 flashGrid(NN / 64, NHD, TT);

    broad_proj_kernel<<<dim3(BB * TT), 256, 0, stream>>>(broadf, Wp, bp, broadb);
    embed_ln_gelu_kernel<<<dim3(NN * TT), 256, 0, stream>>>(nodes, Wn, bn, broadb, types_tab,
                                                            ntypes, gids, g_ip1, b_ip1, X);
    gemm_wmma_kernel<<<gemmGrid, gemmBlk, 0, stream>>>(X, W_ip, b_ip, buf0, MROWS, 0);
    ln_kernel<<<dim3(MROWS), 256, 0, stream>>>(buf0, nullptr, g_ip2, b_ip2, X, 1);

    for (int l = 0; l < LL; ++l) {
        const float* wq = Wq + (size_t)l * HH * HH; const float* bq_ = bq + (size_t)l * HH;
        const float* wk = Wk + (size_t)l * HH * HH; const float* bk_ = bk + (size_t)l * HH;
        const float* wv = Wv + (size_t)l * HH * HH; const float* bv_ = bv + (size_t)l * HH;
        const float* wo = Wo + (size_t)l * HH * HH; const float* bo_ = bo + (size_t)l * HH;

        gemm_wmma_kernel<<<gemmGrid, gemmBlk, 0, stream>>>(X, wq, bq_, buf0, MROWS, 0);
        gemm_wmma_kernel<<<gemmGrid, gemmBlk, 0, stream>>>(X, wk, bk_, buf1, MROWS, 0);
        gemm_wmma_kernel<<<gemmGrid, gemmBlk, 0, stream>>>(X, wv, bv_, buf2, MROWS, 0);
        flash_attn_kernel<<<flashGrid, dim3(128), 0, stream>>>(buf0, buf1, buf2, dist, spat, buf3);
        gemm_wmma_kernel<<<gemmGrid, gemmBlk, 0, stream>>>(buf3, wo, bo_, buf0, MROWS, 0);
        ln_kernel<<<dim3(MROWS), 256, 0, stream>>>(X, buf0,
                                                   ln1g + (size_t)l * HH, ln1b + (size_t)l * HH, X, 0);
        gemm_wmma_kernel<<<gemmGrid, gemmBlk, 0, stream>>>(X, W1f + (size_t)l * HH * HH,
                                                           b1f + (size_t)l * HH, buf1, MROWS, 1);
        gemm_wmma_kernel<<<gemmGrid, gemmBlk, 0, stream>>>(buf1, W2f + (size_t)l * HH * HH,
                                                           b2f + (size_t)l * HH, buf2, MROWS, 0);
        ln_kernel<<<dim3(MROWS), 256, 0, stream>>>(X, buf2,
                                                   ln2g + (size_t)l * HH, ln2b + (size_t)l * HH, X, 0);
    }

    pool_kernel<<<dim3((BB * TT * HH + 255) / 256), 256, 0, stream>>>(X, pooled);
    gemm_wmma_kernel<<<dim3(1, HH / 64), gemmBlk, 0, stream>>>(pooled, Wout, bout, out, BB * TT, 0);
}
